// BoundaryModel_23390391894085
// MI455X (gfx1250) — matrix operand order repaired and checked by reading, not compiled, not run on hardware
//
#include <hip/hip_runtime.h>

#define TT 512
#define WIN 15
#define NEGV -9999999.0f

#if defined(__has_builtin)
#  if __has_builtin(__builtin_amdgcn_tensor_load_to_lds)
#    define HAVE_TDM 1
#  endif
#endif
#ifndef HAVE_TDM
#  define HAVE_TDM 0
#endif
#if defined(__has_include)
#  if __has_include(<hip/amd_detail/amd_gfx1250_TDM.h>)
#    define TDM_SIX_ARGS 1
#  else
#    define TDM_SIX_ARGS 0
#  endif
#else
#  define TDM_SIX_ARGS 0
#endif

typedef __attribute__((ext_vector_type(16))) _Float16 v16h;
typedef __attribute__((ext_vector_type(8)))  float    v8f;
typedef __attribute__((ext_vector_type(4)))  unsigned int v4u;
typedef __attribute__((ext_vector_type(8)))  int      v8i;
typedef __attribute__((ext_vector_type(4)))  int      v4i;

__device__ __forceinline__ v8f wmma16(v16h a, v16h b, v8f c) {
  // D = A(16x32 f16) * B(32x16 f16) + C(16x16 f32)
  return __builtin_amdgcn_wmma_f32_16x16x32_f16(
      /*neg_a=*/false, a, /*neg_b=*/false, b,
      /*c_mod=*/(short)0, c, /*reuse_a=*/false, /*reuse_b=*/false);
}

// fast gate math: single v_exp_f32, no libm special-case branching
__device__ __forceinline__ float sigmoidf(float x) { return 1.0f / (1.0f + __expf(-x)); }
__device__ __forceinline__ float tanh_fast(float x) {
  float e = __expf(2.0f * x);          // +inf -> 1, 0 -> -1 : correct saturation
  return 1.0f - 2.0f / (e + 1.0f);
}

// ---- A fragment: 16x32 f16, row-major source, lane L holds row L%16.
// K pattern (ISA 7.12.2): half h=L/16 -> elems 0..7 = K[h*8 .. h*8+7],
// elems 8..15 = K[16+h*8 .. 16+h*8+7].
__device__ __forceinline__ v16h load_a16(const _Float16* A, int lda, int row0, int k0, int lane) {
  int m = lane & 15, hh = lane >> 4;
  const _Float16* p = A + (size_t)(row0 + m) * lda + k0 + hh * 8;
  v16h a;
#pragma unroll
  for (int i = 0; i < 8; ++i) { a[i] = p[i]; a[8 + i] = p[16 + i]; }
  return a;
}

// A fragment shifted one timestep back (zero row at t==0); rows are (b*T+t).
__device__ __forceinline__ v16h load_a16_shift(const _Float16* A, int lda, int row0, int k0, int lane) {
  int m = lane & 15, hh = lane >> 4;
  int r = row0 + m;
  v16h a;
  if ((r & (TT - 1)) == 0) {
#pragma unroll
    for (int i = 0; i < 16; ++i) a[i] = (_Float16)0.f;
  } else {
    const _Float16* p = A + (size_t)(r - 1) * lda + k0 + hh * 8;
#pragma unroll
    for (int i = 0; i < 8; ++i) { a[i] = p[i]; a[8 + i] = p[16 + i]; }
  }
  return a;
}

// ---- B fragment: 32x16 f16 from weight stored transposed (N,Kpad) row-major.
// Lane L holds column N=L%16; same K pattern as A: elems 0..7 = K[h*8..h*8+7], elems 8..15 = K[16+h*8..16+h*8+7].
__device__ __forceinline__ v16h load_b16(const _Float16* Bt, int ldb, int col0, int k0, int lane) {
  int n = lane & 15, hh = lane >> 4;
  const _Float16* p = Bt + (size_t)(col0 + n) * ldb + k0 + hh * 8;
  v16h b;
#pragma unroll
  for (int i = 0; i < 8; ++i) { b[i] = p[i]; b[8 + i] = p[16 + i]; }
  return b;
}

#if HAVE_TDM
// Issue one TDM 2D tile load: rows x cols f16 tile of a (Ntot x K) row-major
// matrix, tile origin at gsrc, into LDS at byte offset lds_off (compact rows).
// D# layout per CDNA5 ISA.
__device__ __forceinline__ void tdm_load_b_tile(const _Float16* gsrc, unsigned lds_off,
                                                int K, int Ntot, int rows, int cols) {
  unsigned long long ga = (unsigned long long)(size_t)gsrc;
  v4u g0;
  g0[0] = 1u;                                              // count=1 (valid user D#)
  g0[1] = lds_off;                                         // lds_addr (bytes)
  g0[2] = (unsigned)(ga & 0xffffffffu);                    // global_addr[31:0]
  g0[3] = (unsigned)((ga >> 32) & 0x01ffffffu) | (2u << 30); // addr[56:32] | type=2
  v8i g1;
  g1[0] = (int)(1u << 16);                                 // data_size=1 (2 bytes)
  g1[1] = (int)(((unsigned)K & 0xffffu) << 16);            // tensor_dim0[15:0]
  g1[2] = (int)((((unsigned)K >> 16) & 0xffffu) | (((unsigned)Ntot & 0xffffu) << 16));
  g1[3] = (int)((((unsigned)Ntot >> 16) & 0xffffu) | ((unsigned)cols << 16)); // tile_dim0
  g1[4] = (int)((unsigned)rows);                           // tile_dim1 (tile_dim2=0)
  g1[5] = (int)((unsigned)K);                              // tensor_dim0_stride[31:0]
  g1[6] = 0;
  g1[7] = 0;
  v4i z4 = {0, 0, 0, 0};
#if TDM_SIX_ARGS
  v8i z8 = {0, 0, 0, 0, 0, 0, 0, 0};
  __builtin_amdgcn_tensor_load_to_lds(g0, g1, z4, z4, z8, 0);
#else
  __builtin_amdgcn_tensor_load_to_lds(g0, g1, z4, z4, 0);
#endif
}
#endif

// =============== weight transpose/convert: W f32 (K,N) -> Wt f16 (N,Kpad) ===============
__global__ __launch_bounds__(256) void k_wt(const float* __restrict__ Wsrc,
                                            _Float16* __restrict__ Wt,
                                            int K, int N, int Kpad) {
  int idx = blockIdx.x * 256 + threadIdx.x;
  if (idx >= N * Kpad) return;
  int nidx = idx / Kpad, k = idx % Kpad;
  Wt[idx] = (_Float16)((k < K) ? Wsrc[(size_t)k * N + nidx] : 0.f);
}

// =============== embedding gather into x_f16 (8192 x 512 padded) ===============
__global__ __launch_bounds__(256) void k_embed(const int* __restrict__ words,
                                               const int* __restrict__ poss,
                                               const float* __restrict__ wemb,
                                               const float* __restrict__ pemb,
                                               _Float16* __restrict__ x16) {
  int idx = blockIdx.x * 256 + threadIdx.x;
  if (idx >= 8192 * 512) return;
  int row = idx >> 9, col = idx & 511;
  float v;
  if (col < 300)      v = wemb[(size_t)words[row] * 300 + col];
  else if (col < 364) v = pemb[(size_t)poss[row] * 64 + (col - 300)];
  else if (col < 492) return;              // char-LSTM writes these columns
  else                v = 0.f;             // K padding
  x16[idx] = (_Float16)v;
}

// =============== char embedding gather: ce (8192*16, 64) f16 ===============
__global__ __launch_bounds__(256) void k_cembed(const int* __restrict__ chars,
                                                const float* __restrict__ cemb,
                                                _Float16* __restrict__ ce) {
  int idx = blockIdx.x * 256 + threadIdx.x;
  if (idx >= 8192 * 16 * 64) return;
  int c = idx & 63, sl = idx >> 6;
  ce[idx] = (_Float16)cemb[(size_t)chars[sl] * 64 + c];
}

// =============== char LSTM: 16 seqs/block, 8 waves (16 hidden units each) ===============
__global__ __launch_bounds__(256) void k_char_lstm(const _Float16* __restrict__ ce,
                                                   const _Float16* __restrict__ cWi_t,
                                                   const _Float16* __restrict__ cWh_t,
                                                   const float* __restrict__ cb,
                                                   const int* __restrict__ char_len,
                                                   _Float16* __restrict__ x16) {
  __shared__ _Float16 hbuf[2][16][128];
  int lane = threadIdx.x & 31, w = threadIdx.x >> 5;   // wave 0..7
  int u0 = w * 16;
  int s0 = blockIdx.x * 16;
  int n = lane & 15, hh = lane >> 4;
  for (int i = threadIdx.x; i < 16 * 128; i += 256) (&hbuf[0][0][0])[i] = (_Float16)0.f;
  __syncthreads();
  float bg[4];
#pragma unroll
  for (int g = 0; g < 4; ++g) bg[g] = cb[g * 128 + u0 + n];
  v8f cst;
#pragma unroll
  for (int r = 0; r < 8; ++r) cst[r] = 0.f;
  int clen[8];
#pragma unroll
  for (int r = 0; r < 8; ++r) {
    int cl = char_len[s0 + r + 8 * hh] - 1;
    clen[r] = cl < 0 ? 0 : (cl > 15 ? 15 : cl);
  }
  int cur = 0;
  for (int l = 0; l < 16; ++l) {
    v8f acc[4];
#pragma unroll
    for (int g = 0; g < 4; ++g)
#pragma unroll
      for (int r = 0; r < 8; ++r) acc[g][r] = bg[g];
    // x_t @ cWi  (K=64)
    for (int k0 = 0; k0 < 64; k0 += 32) {
      v16h a = load_a16(ce + (size_t)l * 64, 1024, s0, k0, lane);
#pragma unroll
      for (int g = 0; g < 4; ++g)
        acc[g] = wmma16(a, load_b16(cWi_t, 64, g * 128 + u0, k0, lane), acc[g]);
    }
    // h @ cWh  (K=128)
    for (int k0 = 0; k0 < 128; k0 += 32) {
      v16h a = load_a16(&hbuf[cur][0][0], 128, 0, k0, lane);
#pragma unroll
      for (int g = 0; g < 4; ++g)
        acc[g] = wmma16(a, load_b16(cWh_t, 128, g * 128 + u0, k0, lane), acc[g]);
    }
    int nxt = cur ^ 1;
#pragma unroll
    for (int r = 0; r < 8; ++r) {
      float zi = acc[0][r], zf = acc[1][r], zg = acc[2][r], zo = acc[3][r];
      float cn = sigmoidf(zf) * cst[r] + sigmoidf(zi) * tanh_fast(zg);
      cst[r] = cn;
      float hv = sigmoidf(zo) * tanh_fast(cn);
      int sm = r + 8 * hh;
      hbuf[nxt][sm][u0 + n] = (_Float16)hv;
      if (l == clen[r])
        x16[(size_t)(s0 + sm) * 512 + 364 + u0 + n] = (_Float16)hv;
    }
    __syncthreads();
    cur = nxt;
  }
}

// =============== generic GEMM: out = A(M,K)f16 @ Bt(N,K)f16 + bias ===============
// wave -> 16x64 tile; block (8 waves) -> 128x64; grid (N/64, M/128).
// B panel (64 x K) staged through LDS by the Tensor Data Mover, double-buffered
// in 128-element K chunks under TENSORcnt.  Requires K % 128 == 0.
__global__ __launch_bounds__(256) void k_gemm(const _Float16* __restrict__ A, int lda,
                                              const _Float16* __restrict__ Bt, int K,
                                              const float* __restrict__ bias,
                                              float* __restrict__ outF,
                                              _Float16* __restrict__ outH,
                                              int ldn, int relu) {
  int lane = threadIdx.x & 31, w = threadIdx.x >> 5;
  int m0 = blockIdx.y * 128 + w * 16;
  int n0 = blockIdx.x * 64;
  int n = lane & 15, hh = lane >> 4;
  v8f acc[4];
#pragma unroll
  for (int g = 0; g < 4; ++g) {
    float bv = bias ? bias[n0 + g * 16 + n] : 0.f;
#pragma unroll
    for (int r = 0; r < 8; ++r) acc[g][r] = bv;
  }
#if HAVE_TDM
  __shared__ _Float16 bst[2][64 * 128];
  int ntot = gridDim.x * 64;
  int kchunks = K >> 7;
  if (w == 0)
    tdm_load_b_tile(Bt + (size_t)n0 * K, 0u, K, ntot, 64, 128);
  for (int c = 0; c < kchunks; ++c) {
    if (c + 1 < kchunks) {
      if (w == 0) {
        tdm_load_b_tile(Bt + (size_t)n0 * K + (size_t)(c + 1) * 128,
                        (unsigned)(((c + 1) & 1) * (64 * 128 * 2)), K, ntot, 64, 128);
        __builtin_amdgcn_s_wait_tensorcnt(1);   // chunk c landed (in-order)
      }
    } else {
      if (w == 0) __builtin_amdgcn_s_wait_tensorcnt(0);
    }
    __syncthreads();
    const _Float16* bb = &bst[c & 1][0];
    for (int k0 = 0; k0 < 128; k0 += 32) {
      v16h a = load_a16(A, lda, m0, c * 128 + k0, lane);
#pragma unroll
      for (int g = 0; g < 4; ++g)
        acc[g] = wmma16(a, load_b16(bb, 128, g * 16, k0, lane), acc[g]);
    }
    __syncthreads();   // all waves done with buf (c&1) before TDM overwrites it
  }
#else
  for (int k0 = 0; k0 < K; k0 += 32) {
    if (k0 + 32 < K)
      __builtin_prefetch(A + (size_t)(m0 + n) * lda + k0 + 32, 0, 1);
    v16h a = load_a16(A, lda, m0, k0, lane);
#pragma unroll
    for (int g = 0; g < 4; ++g)
      acc[g] = wmma16(a, load_b16(Bt, K, n0 + g * 16, k0, lane), acc[g]);
  }
#endif
#pragma unroll
  for (int g = 0; g < 4; ++g)
#pragma unroll
    for (int r = 0; r < 8; ++r) {
      int row = m0 + r + 8 * hh, col = n0 + g * 16 + n;
      float v = acc[g][r];
      if (relu) v = v > 0.f ? v : 0.f;
      if (outF) outF[(size_t)row * ldn + col] = v;
      if (outH) outH[(size_t)row * ldn + col] = (_Float16)v;
    }
}

// =============== main BiLSTM recurrence: grid=2 (fwd/bwd), 16 waves/block ===============
__global__ __launch_bounds__(512) void k_lstm_main(const float* __restrict__ xiF,
                                                   const float* __restrict__ xiB,
                                                   const _Float16* __restrict__ fWh_t,
                                                   const _Float16* __restrict__ bWh_t,
                                                   const int* __restrict__ seq_len,
                                                   _Float16* __restrict__ hidden) {
  __shared__ _Float16 hbuf[2][16][256];
  int dir = blockIdx.x;
  const float* Xi = dir ? xiB : xiF;
  const _Float16* Wt = dir ? bWh_t : fWh_t;
  int lane = threadIdx.x & 31, w = threadIdx.x >> 5;   // wave 0..15
  int u0 = w * 16;
  int n = lane & 15, hh = lane >> 4;
  for (int i = threadIdx.x; i < 16 * 256; i += 512) (&hbuf[0][0][0])[i] = (_Float16)0.f;
  __syncthreads();
  v8f cst;
#pragma unroll
  for (int r = 0; r < 8; ++r) cst[r] = 0.f;
  int slv[8];
#pragma unroll
  for (int r = 0; r < 8; ++r) slv[r] = seq_len[r + 8 * hh];
  int cur = 0;
  for (int step = 0; step < TT; ++step) {
    int t = dir ? (TT - 1 - step) : step;
    v8f acc[4];
#pragma unroll
    for (int g = 0; g < 4; ++g)
#pragma unroll
      for (int r = 0; r < 8; ++r) {
        int bm = r + 8 * hh;
        acc[g][r] = Xi[((size_t)bm * TT + t) * 1024 + g * 256 + u0 + n];  // x@Wi + b (precomputed)
      }
    for (int k0 = 0; k0 < 256; k0 += 32) {
      v16h a = load_a16(&hbuf[cur][0][0], 256, 0, k0, lane);
#pragma unroll
      for (int g = 0; g < 4; ++g)
        acc[g] = wmma16(a, load_b16(Wt, 256, g * 256 + u0, k0, lane), acc[g]);
    }
    int nxt = cur ^ 1;
#pragma unroll
    for (int r = 0; r < 8; ++r) {
      float zi = acc[0][r], zf = acc[1][r], zg = acc[2][r], zo = acc[3][r];
      float cn = sigmoidf(zf) * cst[r] + sigmoidf(zi) * tanh_fast(zg);
      cst[r] = cn;
      float hv = sigmoidf(zo) * tanh_fast(cn);
      int bm = r + 8 * hh;
      hbuf[nxt][bm][u0 + n] = (_Float16)hv;
      float hm = (t < slv[r]) ? hv : 0.f;                 // seq-len mask on output only
      hidden[((size_t)bm * TT + t) * 512 + dir * 256 + u0 + n] = (_Float16)hm;
    }
    __syncthreads();
    cur = nxt;
  }
}

// =============== time conv: relu(h[t-1]@K0 + h[t]@K1 + b) -> f16 ===============
__global__ __launch_bounds__(256) void k_conv(const _Float16* __restrict__ hid,
                                              const _Float16* __restrict__ W0t,
                                              const _Float16* __restrict__ W1t,
                                              const float* __restrict__ bias,
                                              _Float16* __restrict__ out) {
  int lane = threadIdx.x & 31, w = threadIdx.x >> 5;
  int m0 = blockIdx.y * 128 + w * 16;
  int n0 = blockIdx.x * 64;
  int n = lane & 15, hh = lane >> 4;
  v8f acc[4];
#pragma unroll
  for (int g = 0; g < 4; ++g) {
    float bv = bias[n0 + g * 16 + n];
#pragma unroll
    for (int r = 0; r < 8; ++r) acc[g][r] = bv;
  }
  for (int k0 = 0; k0 < 512; k0 += 32) {
    v16h a = load_a16_shift(hid, 512, m0, k0, lane);
#pragma unroll
    for (int g = 0; g < 4; ++g)
      acc[g] = wmma16(a, load_b16(W0t, 512, n0 + g * 16, k0, lane), acc[g]);
  }
  for (int k0 = 0; k0 < 512; k0 += 32) {
    v16h a = load_a16(hid, 512, m0, k0, lane);
#pragma unroll
    for (int g = 0; g < 4; ++g)
      acc[g] = wmma16(a, load_b16(W1t, 512, n0 + g * 16, k0, lane), acc[g]);
  }
#pragma unroll
  for (int g = 0; g < 4; ++g)
#pragma unroll
    for (int r = 0; r < 8; ++r) {
      int row = m0 + r + 8 * hh, col = n0 + g * 16 + n;
      float v = acc[g][r];
      out[(size_t)row * 512 + col] = (_Float16)(v > 0.f ? v : 0.f);
    }
}

// =============== ll = fl @ w + b  (8192 rows) ===============
__global__ __launch_bounds__(256) void k_lin(const _Float16* __restrict__ f,
                                             const float* __restrict__ wv,
                                             const float* __restrict__ bsc,
                                             float* __restrict__ out) {
  int row = blockIdx.x * 256 + threadIdx.x;
  if (row >= 8192) return;
  float s = bsc[0];
  for (int c = 0; c < 512; ++c) s += (float)f[(size_t)row * 512 + c] * wv[c];
  out[row] = s;
}

// =============== batched bilinear: out[b,i,j]=hl[b,i,:]·fl[b,j,:]+ll[b,j]+mask ===============
__global__ __launch_bounds__(256) void k_bilin(const _Float16* __restrict__ hl,
                                               const _Float16* __restrict__ fl,
                                               const float* __restrict__ llv,
                                               float* __restrict__ out, int is_left) {
  int lane = threadIdx.x & 31, w = threadIdx.x >> 5;
  int b = blockIdx.z;
  int i0 = blockIdx.y * 128 + w * 16;
  int j0 = blockIdx.x * 64;
  int n = lane & 15, hh = lane >> 4;
  const _Float16* Ab = hl + (size_t)b * TT * 512;
  const _Float16* Bb = fl + (size_t)b * TT * 512;
  v8f acc[4];
#pragma unroll
  for (int g = 0; g < 4; ++g)
#pragma unroll
    for (int r = 0; r < 8; ++r) acc[g][r] = 0.f;
  for (int k0 = 0; k0 < 512; k0 += 32) {
    v16h a = load_a16(Ab, 512, i0, k0, lane);
#pragma unroll
    for (int g = 0; g < 4; ++g)
      acc[g] = wmma16(a, load_b16(Bb, 512, j0 + g * 16, k0, lane), acc[g]);
  }
#pragma unroll
  for (int g = 0; g < 4; ++g)
#pragma unroll
    for (int r = 0; r < 8; ++r) {
      int i = i0 + r + 8 * hh, j = j0 + g * 16 + n;
      bool ok = is_left ? (j <= i && j >= i - WIN) : (j >= i && j <= i + WIN);
      out[((size_t)b * TT + i) * TT + j] = acc[g][r] + llv[(size_t)b * TT + j] + (ok ? 0.f : NEGV);
    }
}

// =============== in-place row softmax over 512 (skipped if flag==0) ===============
__global__ __launch_bounds__(256) void k_softmax(float* __restrict__ buf,
                                                 const int* __restrict__ flag) {
  if (flag[0] == 0) return;
  float* p = buf + (size_t)blockIdx.x * TT;
  __shared__ float red[256];
  int tid = threadIdx.x;
  float a = p[tid], b = p[tid + 256];
  red[tid] = fmaxf(a, b);
  __syncthreads();
  for (int s = 128; s > 0; s >>= 1) { if (tid < s) red[tid] = fmaxf(red[tid], red[tid + s]); __syncthreads(); }
  float mx = red[0];
  __syncthreads();
  float e0 = __expf(a - mx), e1 = __expf(b - mx);
  red[tid] = e0 + e1;
  __syncthreads();
  for (int s = 128; s > 0; s >>= 1) { if (tid < s) red[tid] += red[tid + s]; __syncthreads(); }
  float inv = 1.f / red[0];
  p[tid] = e0 * inv;
  p[tid + 256] = e1 * inv;
}

extern "C" void kernel_launch(void* const* d_in, const int* in_sizes, int n_in,
                              void* d_out, int out_size, void* d_ws, size_t ws_size,
                              hipStream_t stream) {
  (void)in_sizes; (void)n_in; (void)out_size;
  char* W = (char*)d_ws;
  size_t off = 0;
  auto take = [&](size_t bytes) -> void* {
    void* p = W + off;
    off = (off + bytes + 255) & ~((size_t)255);
    return p;
  };
  _Float16* x16   = (_Float16*)take(8192ull * 512 * 2);        // concat input, K-padded to 512
  _Float16* ce16  = (_Float16*)take(8192ull * 16 * 64 * 2);    // char embeddings
  float*    xiF   = (float*)take(8192ull * 1024 * 4);          // x@fWi+fb
  float*    xiB   = (float*)take(8192ull * 1024 * 4);          // x@bWi+bb
  _Float16* hid16 = (_Float16*)take(8192ull * 512 * 2);        // masked hidden
  _Float16* fl16  = (_Float16*)take(8192ull * 512 * 2);
  _Float16* fr16  = (_Float16*)take(8192ull * 512 * 2);
  _Float16* hlL   = (_Float16*)take(8192ull * 512 * 2);
  _Float16* hlR   = (_Float16*)take(8192ull * 512 * 2);
  float*    ll    = (float*)take(8192ull * 4);
  float*    lr    = (float*)take(8192ull * 4);
  _Float16* cWi_t = (_Float16*)take(512ull * 64 * 2);
  _Float16* cWh_t = (_Float16*)take(512ull * 128 * 2);
  _Float16* fWi_t = (_Float16*)take(1024ull * 512 * 2);
  _Float16* fWh_t = (_Float16*)take(1024ull * 256 * 2);
  _Float16* bWi_t = (_Float16*)take(1024ull * 512 * 2);
  _Float16* bWh_t = (_Float16*)take(1024ull * 256 * 2);
  _Float16* cl0_t = (_Float16*)take(512ull * 512 * 2);
  _Float16* cl1_t = (_Float16*)take(512ull * 512 * 2);
  _Float16* cr0_t = (_Float16*)take(512ull * 512 * 2);
  _Float16* cr1_t = (_Float16*)take(512ull * 512 * 2);
  _Float16* biL_t = (_Float16*)take(512ull * 512 * 2);
  _Float16* biR_t = (_Float16*)take(512ull * 512 * 2);
  if (off > ws_size) return;  // workspace too small; nothing safe to do

  auto wt = [&](const void* src, _Float16* dst, int K, int N, int Kpad) {
    int tot = N * Kpad;
    k_wt<<<(tot + 255) / 256, 256, 0, stream>>>((const float*)src, dst, K, N, Kpad);
  };
  // weight prep (transpose + f16 convert + K-pad)
  wt(d_in[9],  cWi_t, 64, 512, 64);
  wt(d_in[10], cWh_t, 128, 512, 128);
  wt(d_in[12], fWi_t, 492, 1024, 512);
  wt(d_in[13], fWh_t, 256, 1024, 256);
  wt(d_in[15], bWi_t, 492, 1024, 512);
  wt(d_in[16], bWh_t, 256, 1024, 256);
  wt(d_in[18],                           cl0_t, 512, 512, 512);
  wt((const float*)d_in[18] + 512 * 512, cl1_t, 512, 512, 512);
  wt(d_in[20],                           cr0_t, 512, 512, 512);
  wt((const float*)d_in[20] + 512 * 512, cr1_t, 512, 512, 512);
  wt(d_in[22], biL_t, 512, 512, 512);
  wt(d_in[23], biR_t, 512, 512, 512);

  // embeddings
  k_embed<<<(8192 * 512) / 256, 256, 0, stream>>>(
      (const int*)d_in[0], (const int*)d_in[1],
      (const float*)d_in[6], (const float*)d_in[7], x16);
  k_cembed<<<(8192 * 16 * 64) / 256, 256, 0, stream>>>(
      (const int*)d_in[3], (const float*)d_in[8], ce16);

  // char LSTM -> fills x16 cols [364,492)
  k_char_lstm<<<512, 256, 0, stream>>>(ce16, cWi_t, cWh_t,
                                       (const float*)d_in[11],
                                       (const int*)d_in[4], x16);

  // hoisted input GEMMs: Xi = x @ Wi + b  (M=8192, K=512pad, N=1024)
  k_gemm<<<dim3(16, 64), 256, 0, stream>>>(x16, 512, fWi_t, 512,
                                           (const float*)d_in[14], xiF, nullptr, 1024, 0);
  k_gemm<<<dim3(16, 64), 256, 0, stream>>>(x16, 512, bWi_t, 512,
                                           (const float*)d_in[17], xiB, nullptr, 1024, 0);

  // BiLSTM recurrence (fwd/bwd in parallel workgroups)
  k_lstm_main<<<2, 512, 0, stream>>>(xiF, xiB, fWh_t, bWh_t,
                                     (const int*)d_in[2], hid16);

  // time convolutions
  k_conv<<<dim3(8, 64), 256, 0, stream>>>(hid16, cl0_t, cl1_t, (const float*)d_in[19], fl16);
  k_conv<<<dim3(8, 64), 256, 0, stream>>>(hid16, cr0_t, cr1_t, (const float*)d_in[21], fr16);

  // scalar linears
  k_lin<<<32, 256, 0, stream>>>(fl16, (const float*)d_in[24], (const float*)d_in[25], ll);
  k_lin<<<32, 256, 0, stream>>>(fr16, (const float*)d_in[26], (const float*)d_in[27], lr);

  // hl = hidden @ bilin
  k_gemm<<<dim3(8, 64), 256, 0, stream>>>(hid16, 512, biL_t, 512, nullptr, nullptr, hlL, 512, 0);
  k_gemm<<<dim3(8, 64), 256, 0, stream>>>(hid16, 512, biR_t, 512, nullptr, nullptr, hlR, 512, 0);

  // batched bilinear scores + linear + window mask -> d_out
  float* outL = (float*)d_out;
  float* outR = outL + 16ull * TT * TT;
  k_bilin<<<dim3(8, 4, 16), 256, 0, stream>>>(hlL, fl16, ll, outL, 1);
  k_bilin<<<dim3(8, 4, 16), 256, 0, stream>>>(hlR, fr16, lr, outR, 0);

  // softmax over axis 2 (both halves, 16384 rows of 512), gated by do_softmax
  k_softmax<<<2 * 16 * TT, 256, 0, stream>>>((float*)d_out, (const int*)d_in[5]);
}
